// NDConv_53429393162448
// MI455X (gfx1250) — compile-verified, matrix operand order checked by reading, not run on hardware
//
#include <hip/hip_runtime.h>
#include <hip/hip_bf16.h>
#include <stdint.h>

// CDNA5 / gfx1250: wave32, WMMA 16x16x32 bf16 -> f32 implicit-GEMM 3x3 conv.
typedef __attribute__((ext_vector_type(16))) __bf16 v16bf;
typedef __attribute__((ext_vector_type(8)))  float  v8f;

#define TILE_N   128            // output pixels (x) per block
#define RSTRIDE  72             // bf16 elems per LDS row (64 cins + 8 pad -> conflict-free b128)
#define A_SLAB   (64 * RSTRIDE) // one (ky,kx) weight slab: 64 couts x 64 cins
#define B_ROWS   130            // TILE_N + 2 halo pixels for the kx shifts
#define B_SLAB   (B_ROWS * RSTRIDE)
#define AW_ELEMS (9 * A_SLAB)                    // all 9 weight slabs
#define LDS_ELEMS (AW_ELEMS + 2 * B_SLAB)        // + double-buffered B strip
#define LDS_BYTES (LDS_ELEMS * 2)                // 120,384 B of the 320 KB WGP LDS

// packed f32 -> bf16 pair (RNE); HW packed convert if the toolchain grows one
__device__ __forceinline__ uint32_t pack2_bf16(float a, float b) {
#if __has_builtin(__builtin_amdgcn_cvt_pk_bf16_f32)
    auto p = __builtin_amdgcn_cvt_pk_bf16_f32(a, b);
    return __builtin_bit_cast(uint32_t, p);
#else
    uint32_t ua = __builtin_bit_cast(uint32_t, a);
    uint32_t ub = __builtin_bit_cast(uint32_t, b);
    ua += 0x7FFFu + ((ua >> 16) & 1u);
    ub += 0x7FFFu + ((ub >> 16) & 1u);
    return (ua >> 16) | (ub & 0xFFFF0000u);
#endif
}

__global__ __launch_bounds__(256) void conv3x3_wmma_bf16(
    const float* __restrict__ x,     // [8,64,256,256]
    const float* __restrict__ w,     // [64,64,3,3]
    const float* __restrict__ bias,  // [64]
    float* __restrict__ out)         // [8,64,254,254]
{
    extern __shared__ __bf16 smem[];
    __bf16* Aw = smem;                                   // [9][64][RSTRIDE]
    __bf16* Bb0 = smem + AW_ELEMS;                       // [B_ROWS][RSTRIDE] x2
    __bf16* Bb1 = smem + AW_ELEMS + B_SLAB;

    const int tid  = threadIdx.x;
    const int lane = tid & 31;
    const int wave = tid >> 5;       // 0..7
    const int mt   = wave & 3;       // M tile: couts mt*16 .. +15
    const int nh   = wave >> 2;      // N half: pixels nh*64 .. +63
    const int half = lane >> 4;      // K-half selector per ISA bf16 layout
    const int l16  = lane & 15;

    const int x0 = blockIdx.x * TILE_N;   // 0 or 128
    const int y0 = blockIdx.y;            // output row 0..253
    const int b  = blockIdx.z;            // batch

    // ---- one-time: all 9 weight slabs -> LDS (converted once per block) ----
    #pragma unroll 4
    for (int i = 0; i < 72; ++i) {
        int idx = tid + i * 256;          // 0..18431 exactly
        int cin = (idx & 31) * 2;         // 0,2,..,62
        int m   = (idx >> 5) & 63;
        int pos = idx >> 11;              // 0..8
        const float* wp = w + ((m * 64 + cin) * 9) + pos;
        *(uint32_t*)&Aw[pos * A_SLAB + m * RSTRIDE + cin] = pack2_bf16(wp[0], wp[9]);
    }

    // ---- B strip fill: one input row (all 64 cins, 130 pixels) per ky ------
    // Pixel-major layout: the kx tap is a +kx row shift at fragment-load time,
    // so one fill + one convert serves all 3 kx and both cin halves.
    auto fillB = [&](__bf16* buf, int ky) {
        #pragma unroll 4
        for (int i = 0; i < 17; ++i) {
            int idx = tid + i * 256;
            if (idx < B_ROWS * 32) {          // 4160 pair-slots
                int n  = idx % B_ROWS;        // coalesced: consecutive lanes -> consecutive xi
                int c  = (idx / B_ROWS) * 2;  // cin pair
                int xi = x0 + n; if (xi > 255) xi = 255;   // halo feeds only guarded outputs
                const float* xp = x + (((size_t)(b * 64 + c) * 256) + (size_t)(y0 + ky)) * 256 + xi;
                *(uint32_t*)&buf[n * RSTRIDE + c] = pack2_bf16(xp[0], xp[256 * 256]);
            }
        }
    };

    v8f acc[4];
    #pragma unroll
    for (int j = 0; j < 4; ++j) acc[j] = (v8f){0.f,0.f,0.f,0.f,0.f,0.f,0.f,0.f};

    fillB(Bb0, 0);   // prologue

    for (int ky = 0; ky < 3; ++ky) {
        __syncthreads();                       // strip ky staged; strip ky-1 readers done
        if (ky + 1 < 3) fillB((ky & 1) ? Bb0 : Bb1, ky + 1);   // overlap with 24 WMMAs
        const __bf16* Bbuf = (ky & 1) ? Bb1 : Bb0;

        #pragma unroll
        for (int kx = 0; kx < 3; ++kx) {
            const __bf16* As = Aw + (ky * 3 + kx) * A_SLAB;
            #pragma unroll
            for (int cc = 0; cc < 2; ++cc) {
                const int kofs = cc * 32 + half * 8;   // lanes<16: K{0..7,16..23}; lanes>=16: K{8..15,24..31}
                union { v16bf v; uint4 q[2]; } af, bf[4];
                {
                    int base = (mt * 16 + l16) * RSTRIDE + kofs;
                    af.q[0] = *(const uint4*)&As[base];
                    af.q[1] = *(const uint4*)&As[base + 16];
                }
                #pragma unroll
                for (int j = 0; j < 4; ++j) {          // cluster loads, then WMMAs
                    int base = (nh * 64 + j * 16 + l16 + kx) * RSTRIDE + kofs;
                    bf[j].q[0] = *(const uint4*)&Bbuf[base];
                    bf[j].q[1] = *(const uint4*)&Bbuf[base + 16];
                }
                #pragma unroll
                for (int j = 0; j < 4; ++j)
                    acc[j] = __builtin_amdgcn_wmma_f32_16x16x32_bf16(
                        false, af.v, false, bf[j].v, (short)0, acc[j], false, false);
            }
        }
    }

    // Epilogue: VGPR r holds M=r (lanes 0-15) / M=r+8 (lanes 16-31), N=lane%16.
    #pragma unroll
    for (int j = 0; j < 4; ++j) {
        int xo = x0 + nh * 64 + j * 16 + l16;
        if (xo < 254) {
            #pragma unroll
            for (int r = 0; r < 8; ++r) {
                int cout = mt * 16 + half * 8 + r;
                size_t oidx = (((size_t)b * 64 + cout) * 254 + y0) * 254 + xo;
                out[oidx] = acc[j][r] + bias[cout];
            }
        }
    }
}

extern "C" void kernel_launch(void* const* d_in, const int* in_sizes, int n_in,
                              void* d_out, int out_size, void* d_ws, size_t ws_size,
                              hipStream_t stream) {
    const float* x    = (const float*)d_in[0];
    const float* w    = (const float*)d_in[1];
    const float* bias = (const float*)d_in[2];
    float* out        = (float*)d_out;

    dim3 grid((254 + TILE_N - 1) / TILE_N, 254, 8);  // (2, 254, 8)
    dim3 block(256);                                  // 8 wave32 waves
    conv3x3_wmma_bf16<<<grid, block, LDS_BYTES, stream>>>(x, w, bias, out);
}
